// CIN_51634096833191
// MI455X (gfx1250) — compile-verified, matrix operand order checked by reading, not run on hardware
//
#include <hip/hip_runtime.h>

#define F0N 39
#define NB  4096
#define DD  16
#define SSN 128

typedef __attribute__((ext_vector_type(8)))  float  v8f;
typedef __attribute__((ext_vector_type(16))) __bf16 v16bf;

union BF16U { __bf16 h; unsigned short u; };
union AFrag { v16bf v; __bf16 e[16]; };

__device__ __forceinline__ unsigned short f2bf(float f) {
  BF16U t; t.h = (__bf16)f; return t.u;
}

// Pack W (f32, row-major [F0*fk, 128]) into WMMA B-fragment order (bf16):
// wpre[(((i*nch + c)*8 + nt)*32 + l)*16 + p] =
//   bf16( W[(i*fk + c*32 + ks(l,p))*128 + nt*16 + (l&15)] ), zero-padded j>=fk.
// ks(l,p): lanes 0-15 hold K {0..7,16..23}, lanes 16-31 hold K {8..15,24..31}.
__global__ void pack_w_kernel(const float* __restrict__ w,
                              unsigned short* __restrict__ wpre,
                              int fk, int nch) {
  int total = F0N * nch * 8 * 32 * 16;
  for (int e = blockIdx.x * blockDim.x + threadIdx.x; e < total;
       e += gridDim.x * blockDim.x) {
    int p    = e & 15;
    int l    = (e >> 4) & 31;
    int nt   = (e >> 9) & 7;
    int rest = e >> 12;                 // i*nch + c
    int c    = rest % nch;
    int i    = rest / nch;
    int ks   = (p < 8 ? p : p + 8) + (l < 16 ? 0 : 8);
    int j    = c * 32 + ks;
    int s    = nt * 16 + (l & 15);
    float v  = (j < fk) ? w[((size_t)i * fk + j) * SSN + s] : 0.0f;
    wpre[e]  = f2bf(v);
  }
}

// One CIN layer's K-loop: M-tiles = 2 batch rows per wave, N = 128 (8 tiles),
// K = i*fkpad + j. A generated on the fly: scalar(x0[b,i,d]) * xk[b,j,d].
// B-fragments batch-loaded (8 per chunk, one wait) and each feeds 2 WMMAs.
template<int NCH>
__device__ __forceinline__ void run_layer(
    const unsigned short* __restrict__ wpre, int lane, int half,
    const float* __restrict__ xr0a, const float* __restrict__ xr0b,
    const float* __restrict__ xrka, const float* __restrict__ xrkb,
    v8f (&acc)[2][8])
{
  const v16bf* wpl = (const v16bf*)wpre + lane;
  for (int i = 0; i < F0N; ++i) {
    if (i + 1 < F0N)   // pull next i-block of B-fragments toward the WGP
      __builtin_prefetch(wpre + ((size_t)(i + 1) * NCH * 256 + lane) * 16, 0, 0);
    const float sA = xr0a[i];
    const float sB = xr0b[i];
#pragma unroll
    for (int c = 0; c < NCH; ++c) {
      const v16bf* wp = wpl + (size_t)(i * NCH + c) * 256;
      v16bf bf[8];
#pragma unroll
      for (int nt = 0; nt < 8; ++nt) bf[nt] = wp[nt * 32];

      const int base = c * 32 + half * 8;
      float4 q0 = *(const float4*)&xrka[base + 0];
      float4 q1 = *(const float4*)&xrka[base + 4];
      float4 q2 = *(const float4*)&xrka[base + 16];
      float4 q3 = *(const float4*)&xrka[base + 20];
      AFrag afA;
      afA.e[ 0] = (__bf16)(sA * q0.x); afA.e[ 1] = (__bf16)(sA * q0.y);
      afA.e[ 2] = (__bf16)(sA * q0.z); afA.e[ 3] = (__bf16)(sA * q0.w);
      afA.e[ 4] = (__bf16)(sA * q1.x); afA.e[ 5] = (__bf16)(sA * q1.y);
      afA.e[ 6] = (__bf16)(sA * q1.z); afA.e[ 7] = (__bf16)(sA * q1.w);
      afA.e[ 8] = (__bf16)(sA * q2.x); afA.e[ 9] = (__bf16)(sA * q2.y);
      afA.e[10] = (__bf16)(sA * q2.z); afA.e[11] = (__bf16)(sA * q2.w);
      afA.e[12] = (__bf16)(sA * q3.x); afA.e[13] = (__bf16)(sA * q3.y);
      afA.e[14] = (__bf16)(sA * q3.z); afA.e[15] = (__bf16)(sA * q3.w);

      q0 = *(const float4*)&xrkb[base + 0];
      q1 = *(const float4*)&xrkb[base + 4];
      q2 = *(const float4*)&xrkb[base + 16];
      q3 = *(const float4*)&xrkb[base + 20];
      AFrag afB;
      afB.e[ 0] = (__bf16)(sB * q0.x); afB.e[ 1] = (__bf16)(sB * q0.y);
      afB.e[ 2] = (__bf16)(sB * q0.z); afB.e[ 3] = (__bf16)(sB * q0.w);
      afB.e[ 4] = (__bf16)(sB * q1.x); afB.e[ 5] = (__bf16)(sB * q1.y);
      afB.e[ 6] = (__bf16)(sB * q1.z); afB.e[ 7] = (__bf16)(sB * q1.w);
      afB.e[ 8] = (__bf16)(sB * q2.x); afB.e[ 9] = (__bf16)(sB * q2.y);
      afB.e[10] = (__bf16)(sB * q2.z); afB.e[11] = (__bf16)(sB * q2.w);
      afB.e[12] = (__bf16)(sB * q3.x); afB.e[13] = (__bf16)(sB * q3.y);
      afB.e[14] = (__bf16)(sB * q3.z); afB.e[15] = (__bf16)(sB * q3.w);

#pragma unroll
      for (int nt = 0; nt < 8; ++nt) {
        acc[0][nt] = __builtin_amdgcn_wmma_f32_16x16x32_bf16(
            false, afA.v, false, bf[nt], (short)0, acc[0][nt], false, false);
        acc[1][nt] = __builtin_amdgcn_wmma_f32_16x16x32_bf16(
            false, afB.v, false, bf[nt], (short)0, acc[1][nt], false, false);
      }
    }
  }
}

// Fully fused 3-layer CIN: per-b independence lets the whole chain run in one
// kernel with intermediate activations (x1, x2) living only in LDS.
// No inter-layer barriers needed: each wave reads only rows of its own
// b-pair that it wrote itself (per-wave LDS ops are in-order).
__global__ void __launch_bounds__(64) cin_fused_kernel(
    const float* __restrict__ xg,
    const unsigned short* __restrict__ wp0, const float* __restrict__ bias0,
    const unsigned short* __restrict__ wp1, const float* __restrict__ bias1,
    const unsigned short* __restrict__ wp2, const float* __restrict__ bias2,
    float* __restrict__ out)
{
  __shared__ float lds_x0[4][16][40];     // [b][d][i], padded
  __shared__ float lds_xk[4][16][132];    // [b][d][j], padded; reused per layer

  const int tid = threadIdx.x;
  const int b0  = blockIdx.x * 4;

  // Stage x tile transposed into lds_x0 AND into lds_xk (layer-0 input,
  // zero-padded to j<64); float4 over d.
  for (int e = tid; e < 4 * 64 * 4; e += 64) {
    int bl = e / (64 * 4);
    int r  = e - bl * (64 * 4);
    int i  = r >> 2, d4 = (r & 3) * 4;
    float4 v = make_float4(0.f, 0.f, 0.f, 0.f);
    if (i < F0N)
      v = *(const float4*)&xg[((size_t)(b0 + bl) * F0N + i) * DD + d4];
    if (i < F0N) {
      lds_x0[bl][d4 + 0][i] = v.x; lds_x0[bl][d4 + 1][i] = v.y;
      lds_x0[bl][d4 + 2][i] = v.z; lds_x0[bl][d4 + 3][i] = v.w;
    }
    lds_xk[bl][d4 + 0][i] = v.x; lds_xk[bl][d4 + 1][i] = v.y;
    lds_xk[bl][d4 + 2][i] = v.z; lds_xk[bl][d4 + 3][i] = v.w;
  }
  __syncthreads();

  const int wave = tid >> 5;      // wave owns b-pair {wave*2, wave*2+1}
  const int lane = tid & 31;
  const int d    = lane & 15;     // A-operand row (=d); C-operand column
  const int half = lane >> 4;

  const float* xr0a = &lds_x0[wave * 2 + 0][d][0];
  const float* xr0b = &lds_x0[wave * 2 + 1][d][0];
  const float* xrka = &lds_xk[wave * 2 + 0][d][0];
  const float* xrkb = &lds_xk[wave * 2 + 1][d][0];

  // ---------------- Layer 0: K = 39 x 64(padded), x1 -> LDS only ----------
  {
    v8f acc[2][8] = {};
    run_layer<2>(wp0, lane, half, xr0a, xr0b, xrka, xrkb, acc);
#pragma unroll
    for (int u = 0; u < 2; ++u) {
#pragma unroll
      for (int nt = 0; nt < 8; ++nt) {
        int s = nt * 16 + d;
        float bs = bias0[s];
#pragma unroll
        for (int r = 0; r < 8; ++r)
          lds_xk[wave * 2 + u][half * 8 + r][s] = acc[u][nt][r] + bs;
      }
    }
  }

  // ---------------- Layer 1: K = 39 x 128, x2 -> LDS, sum_d -> out[:,0:128]
  {
    v8f acc[2][8] = {};
    run_layer<4>(wp1, lane, half, xr0a, xr0b, xrka, xrkb, acc);
#pragma unroll
    for (int u = 0; u < 2; ++u) {
      const int b = b0 + wave * 2 + u;
#pragma unroll
      for (int nt = 0; nt < 8; ++nt) {
        int s = nt * 16 + d;
        float bs = bias1[s];
        float part = 0.0f;
#pragma unroll
        for (int r = 0; r < 8; ++r) part += acc[u][nt][r];
        float other = __shfl(part, lane ^ 16, 32);
        if (half == 0)
          out[(size_t)b * 256 + s] = part + other + 16.0f * bs;
      }
    }
    // write x2 into LDS after the sum (all K-reads of layer 1 are done;
    // per-wave LDS ordering makes this safe for layer 2's reads)
#pragma unroll
    for (int u = 0; u < 2; ++u) {
#pragma unroll
      for (int nt = 0; nt < 8; ++nt) {
        int s = nt * 16 + d;
        float bs = bias1[s];
#pragma unroll
        for (int r = 0; r < 8; ++r)
          lds_xk[wave * 2 + u][half * 8 + r][s] = acc[u][nt][r] + bs;
      }
    }
  }

  // ---------------- Layer 2: K = 39 x 128, sum_d -> out[:,128:256] --------
  {
    v8f acc[2][8] = {};
    run_layer<4>(wp2, lane, half, xr0a, xr0b, xrka, xrkb, acc);
#pragma unroll
    for (int u = 0; u < 2; ++u) {
      const int b = b0 + wave * 2 + u;
#pragma unroll
      for (int nt = 0; nt < 8; ++nt) {
        int s = nt * 16 + d;
        float part = 0.0f;
#pragma unroll
        for (int r = 0; r < 8; ++r) part += acc[u][nt][r];
        float other = __shfl(part, lane ^ 16, 32);
        if (half == 0)
          out[(size_t)b * 256 + 128 + s] = part + other + 16.0f * bias2[s];
      }
    }
  }
}

extern "C" void kernel_launch(void* const* d_in, const int* in_sizes, int n_in,
                              void* d_out, int out_size, void* d_ws, size_t ws_size,
                              hipStream_t stream) {
  (void)in_sizes; (void)n_in; (void)out_size; (void)ws_size;
  const float* x  = (const float*)d_in[0];
  const float* w0 = (const float*)d_in[1];
  const float* b0 = (const float*)d_in[2];
  const float* w1 = (const float*)d_in[3];
  const float* b1 = (const float*)d_in[4];
  const float* w2 = (const float*)d_in[5];
  const float* b2 = (const float*)d_in[6];
  float* out = (float*)d_out;

  char* ws = (char*)d_ws;
  unsigned short* wp0 = (unsigned short*)(ws);                 // 0.61 MB
  unsigned short* wp1 = (unsigned short*)(ws + (2u << 20));    // 1.22 MB
  unsigned short* wp2 = (unsigned short*)(ws + (4u << 20));    // 1.22 MB

  // Pre-pack weights into WMMA B-fragment order (L2-resident, reused 1024x)
  pack_w_kernel<<<256, 256, 0, stream>>>(w0, wp0, 39, 2);
  pack_w_kernel<<<256, 256, 0, stream>>>(w1, wp1, 128, 4);
  pack_w_kernel<<<256, 256, 0, stream>>>(w2, wp2, 128, 4);

  // Single fused kernel: all three CIN layers, intermediates in LDS only.
  cin_fused_kernel<<<NB / 4, 64, 0, stream>>>(x, wp0, b0, wp1, b1,
                                              wp2, b2, out);
}
